// LayeredLSTM_27513560498939
// MI455X (gfx1250) — matrix operand order repaired and checked by reading, not compiled, not run on hardware
//
#include <hip/hip_runtime.h>

#define X_IN   47508
#define HH     128
#define LL     16
#define BSZ    128
#define XTOT   47764          // X_IN + 2*H
#define K1     47636          // X_IN + H   (layer-independent K range)
#define VT     3620
#define NSTEPS 1489           // ceil(K1/32)
#define KC     8
#define SPC    187            // steps per K-chunk (8*187 >= 1489)
#define NCT    8              // 512/64 column tiles

typedef __attribute__((ext_vector_type(16))) __bf16 v16bf;
typedef __attribute__((ext_vector_type(8)))  float  v8f;

union Frag16 { v16bf v; __bf16 h[16]; };

static __device__ __forceinline__ v8f wmma_bf16(const Frag16& a, const Frag16& b, v8f c) {
    return __builtin_amdgcn_wmma_f32_16x16x32_bf16(false, a.v, false, b.v, (short)0, c, false, false);
}

static __device__ __forceinline__ float sigm(float x) { return 1.0f / (1.0f + __expf(-x)); }

// ---------------------------------------------------------------------------
// Phase 1: partial[kc][l][b][c] = (input_x ++ prev_hidden[:,l]) @ Wg[l][0:K1, c]
// Grid: 16(L) * 8(colTile) * 8(KC) blocks, 256 threads (8 waves).
// Wave = one 16x64 output strip: 1 A fragment reused across 4 B fragments/step.
// ---------------------------------------------------------------------------
__global__ __launch_bounds__(256) void k_gemm_big(
    const float* __restrict__ x, const float* __restrict__ prevh,
    const float* __restrict__ Wi, const float* __restrict__ Wf,
    const float* __restrict__ Ws, const float* __restrict__ Wo,
    float* __restrict__ part)
{
    const int bid = blockIdx.x;
    const int kc  = bid & 7;
    const int ct  = (bid >> 3) & 7;
    const int l   = bid >> 6;
    const int lane = threadIdx.x & 31;
    const int wv   = threadIdx.x >> 5;

    const int mBase  = wv * 16;
    const int row    = mBase + (lane & 15);
    const int kbA    = (lane >> 4) * 8;       // A K-offset per lane-half
    const int kbB    = (lane >> 4) * 8;       // B K-offset per lane-half
    const int colBlk = ct * 64;               // 64-col strip, never crosses a gate
    const int gate   = ct >> 1;               // gate order: i, f, s, o
    const float* W   = (gate == 0 ? Wi : (gate == 1 ? Wf : (gate == 2 ? Ws : Wo)))
                       + (size_t)l * XTOT * HH;
    const int bc     = (colBlk & 127) + (lane & 15);

    const float* xrow = x + (size_t)row * X_IN;
    const float* hrow = prevh + ((size_t)row * LL + l) * HH;

    v8f acc0 = {}, acc1 = {}, acc2 = {}, acc3 = {};

    const int sBeg = kc * SPC;
    const int sEnd = (sBeg + SPC < NSTEPS) ? (sBeg + SPC) : NSTEPS;

    for (int s = sBeg; s < sEnd; ++s) {
        const int k0 = s * 32;
        Frag16 a;
        const int ka = k0 + kbA;
        if (k0 + 31 < X_IN) {                       // uniform fast path: pure input_x
            const float4* p = reinterpret_cast<const float4*>(xrow + ka);
            float t[16];
            *reinterpret_cast<float4*>(t + 0)  = p[0];
            *reinterpret_cast<float4*>(t + 4)  = p[1];
            *reinterpret_cast<float4*>(t + 8)  = p[4];   // K = ka+16..19
            *reinterpret_cast<float4*>(t + 12) = p[5];   // K = ka+20..23
            #pragma unroll
            for (int e = 0; e < 16; ++e) a.h[e] = (__bf16)t[e];
        } else {                                    // boundary: input_x | prev_hidden | pad
            #pragma unroll
            for (int e = 0; e < 16; ++e) {
                const int k = ka + (e & 7) + ((e >> 3) << 4);
                float v = 0.0f;
                if (k < X_IN)      v = xrow[k];
                else if (k < K1)   v = hrow[k - X_IN];
                a.h[e] = (__bf16)v;
            }
        }
        #pragma unroll
        for (int n = 0; n < 4; ++n) {
            Frag16 b;
            #pragma unroll
            for (int e = 0; e < 16; ++e)
                b.h[e] = (__bf16)W[(size_t)(k0 + kbB + (e & 7) + ((e >> 3) << 4)) * HH + bc + n * 16];
            if      (n == 0) acc0 = wmma_bf16(a, b, acc0);
            else if (n == 1) acc1 = wmma_bf16(a, b, acc1);
            else if (n == 2) acc2 = wmma_bf16(a, b, acc2);
            else             acc3 = wmma_bf16(a, b, acc3);
        }
    }

    float* pout = part + ((size_t)kc * LL + l) * BSZ * 512;
    const int rOff = (lane >> 4) << 3;
    const int cL   = lane & 15;
    #pragma unroll
    for (int r = 0; r < 8; ++r) {
        const size_t ro = (size_t)(mBase + r + rOff) * 512;
        pout[ro + colBlk +  0 + cL] = acc0[r];
        pout[ro + colBlk + 16 + cL] = acc1[r];
        pout[ro + colBlk + 32 + cL] = acc2[r];
        pout[ro + colBlk + 48 + cL] = acc3[r];
    }
}

// ---------------------------------------------------------------------------
// Deterministic split-K reduction + bias:  pre = sum_kc part[kc] + bias
// ---------------------------------------------------------------------------
__global__ __launch_bounds__(256) void k_reduce(
    const float* __restrict__ part,
    const float* __restrict__ bi, const float* __restrict__ bf_,
    const float* __restrict__ bs, const float* __restrict__ bo,
    float* __restrict__ pre)
{
    const size_t i = (size_t)blockIdx.x * 256 + threadIdx.x;   // < 16*128*512
    float s = 0.0f;
    #pragma unroll
    for (int kc = 0; kc < KC; ++kc) s += part[(size_t)kc * (LL * BSZ * 512) + i];
    const int c = (int)(i & 511);
    const int l = (int)(i >> 16);
    const int h = c & 127;
    const int g = c >> 7;
    const float* bias = (g == 0 ? bi : (g == 1 ? bf_ : (g == 2 ? bs : bo)));
    pre[i] = s + bias[l * HH + h];
}

// ---------------------------------------------------------------------------
// Phase 2a (per layer l>=1): pre[l] += hidden[l-1] @ Wg[l][K1:K1+128, :]
// Grid: 32 blocks (N tiles) * 8 waves (M tiles), K=128.
// ---------------------------------------------------------------------------
__global__ __launch_bounds__(256) void k_delta(
    const float* __restrict__ hid,
    const float* __restrict__ Wi, const float* __restrict__ Wf,
    const float* __restrict__ Ws, const float* __restrict__ Wo,
    float* __restrict__ pre, int l)
{
    const int lane = threadIdx.x & 31;
    const int m    = threadIdx.x >> 5;
    const int nt   = blockIdx.x;              // 0..31
    const int colBase = nt * 16;
    const int gate    = colBase >> 7;
    const float* W = (gate == 0 ? Wi : (gate == 1 ? Wf : (gate == 2 ? Ws : Wo)))
                     + (size_t)l * XTOT * HH;
    const int wcol = (colBase & 127) + (lane & 15);
    const int row  = m * 16 + (lane & 15);
    const int kbA  = (lane >> 4) * 8;
    const int kbB  = (lane >> 4) * 8;
    const float* arow = hid + ((size_t)(l - 1) * BSZ + row) * HH;

    v8f acc = {};
    #pragma unroll
    for (int k0 = 0; k0 < 128; k0 += 32) {
        Frag16 a, b;
        #pragma unroll
        for (int e = 0; e < 16; ++e)
            a.h[e] = (__bf16)arow[k0 + kbA + (e & 7) + ((e >> 3) << 4)];
        #pragma unroll
        for (int e = 0; e < 16; ++e)
            b.h[e] = (__bf16)W[(size_t)(K1 + k0 + kbB + (e & 7) + ((e >> 3) << 4)) * HH + wcol];
        acc = wmma_bf16(a, b, acc);
    }
    float* p = pre + (size_t)l * BSZ * 512;
    const int rOff = (lane >> 4) << 3;
    #pragma unroll
    for (int r = 0; r < 8; ++r)
        p[(size_t)(m * 16 + r + rOff) * 512 + colBase + (lane & 15)] += acc[r];
}

// ---------------------------------------------------------------------------
// Phase 2b (per layer): gates -> new hidden
// ---------------------------------------------------------------------------
__global__ __launch_bounds__(256) void k_act(
    const float* __restrict__ pre, const float* __restrict__ old_states,
    float* __restrict__ hid, int l)
{
    const int i = blockIdx.x * 256 + threadIdx.x;  // < 128*128
    const int b = i >> 7, h = i & 127;
    const float* g = pre + ((size_t)l * BSZ + b) * 512;
    const float ig = sigm(g[h]);
    const float fg = sigm(g[128 + h]);
    const float sg = tanhf(g[256 + h]);
    const float og = sigm(g[384 + h]);
    const float os = old_states[((size_t)l * BSZ + b) * HH + h];
    const float ns = fg * os + ig * sg;
    hid[((size_t)l * BSZ + b) * HH + h] = og * tanhf(ns);
}

// ---------------------------------------------------------------------------
// Phase 3: out[128,3620] = flat_hidden[128,2048] @ Wy[2048,3620]
// Grid: 227 blocks (N tiles, last partial) * 8 waves (M tiles), K=2048.
// ---------------------------------------------------------------------------
__global__ __launch_bounds__(256) void k_out(
    const float* __restrict__ hid, const float* __restrict__ Wy,
    float* __restrict__ out)
{
    const int lane = threadIdx.x & 31;
    const int m    = threadIdx.x >> 5;
    const int colBase = blockIdx.x * 16;
    const int row  = m * 16 + (lane & 15);
    const int kbA  = (lane >> 4) * 8;
    const int kbB  = (lane >> 4) * 8;
    const int col  = colBase + (lane & 15);
    const bool cok = col < VT;
    const int ccol = cok ? col : 0;

    v8f acc = {};
    for (int k0 = 0; k0 < LL * HH; k0 += 32) {
        Frag16 a, b;
        #pragma unroll
        for (int e = 0; e < 16; ++e) {
            const int k = k0 + kbA + (e & 7) + ((e >> 3) << 4);   // k = l*128 + h
            a.h[e] = (__bf16)hid[((size_t)(k >> 7) * BSZ + row) * HH + (k & 127)];
        }
        #pragma unroll
        for (int e = 0; e < 16; ++e)
            b.h[e] = (__bf16)Wy[(size_t)(k0 + kbB + (e & 7) + ((e >> 3) << 4)) * VT + ccol];
        acc = wmma_bf16(a, b, acc);
    }
    const int rOff = (lane >> 4) << 3;
    #pragma unroll
    for (int r = 0; r < 8; ++r)
        if (cok) out[(size_t)(m * 16 + r + rOff) * VT + col] = acc[r];
}

// ---------------------------------------------------------------------------
extern "C" void kernel_launch(void* const* d_in, const int* in_sizes, int n_in,
                              void* d_out, int out_size, void* d_ws, size_t ws_size,
                              hipStream_t stream) {
    const float* x     = (const float*)d_in[0];
    const float* prevh = (const float*)d_in[1];
    const float* oldst = (const float*)d_in[2];
    const float* Wi    = (const float*)d_in[3];
    const float* bi    = (const float*)d_in[4];
    const float* Wf    = (const float*)d_in[5];
    const float* bf_   = (const float*)d_in[6];
    const float* Wo    = (const float*)d_in[7];
    const float* bo    = (const float*)d_in[8];
    const float* Ws    = (const float*)d_in[9];
    const float* bs    = (const float*)d_in[10];
    const float* Wy    = (const float*)d_in[11];
    float* out = (float*)d_out;

    char* ws = (char*)d_ws;
    float* part = (float*)ws;                                    // 8*16*128*512*4  = 32 MiB
    float* pre  = (float*)(ws + (size_t)KC * LL * BSZ * 512 * 4);// 16*128*512*4    =  4 MiB
    float* hid  = (float*)(ws + (size_t)(KC + 1) * LL * BSZ * 512 * 4); // 16*128*128*4 = 1 MiB

    k_gemm_big<<<LL * NCT * KC, 256, 0, stream>>>(x, prevh, Wi, Wf, Ws, Wo, part);
    k_reduce<<<(LL * BSZ * 512) / 256, 256, 0, stream>>>(part, bi, bf_, bs, bo, pre);
    for (int l = 0; l < LL; ++l) {
        if (l > 0) k_delta<<<32, 256, 0, stream>>>(hid, Wi, Wf, Ws, Wo, pre, l);
        k_act<<<(BSZ * HH) / 256, 256, 0, stream>>>(pre, oldst, hid, l);
    }
    k_out<<<(VT + 15) / 16, 256, 0, stream>>>(hid, Wy, out);
}
